// Attention_58798102282639
// MI455X (gfx1250) — matrix operand order repaired and checked by reading, not compiled, not run on hardware
//
#include <hip/hip_runtime.h>

typedef __attribute__((ext_vector_type(16))) __bf16 v16bf;
typedef __attribute__((ext_vector_type(8)))  __bf16 v8bf;
typedef __attribute__((ext_vector_type(8)))  float  v8f;

#define SCALE 0.03608439182435161f   // 768^-0.5
#define DMODEL 768
#define SEQ 4096
#define QT 32
#define KT 64

#define USE_ASYNC_LDS 1

// ---------------- async global->LDS (CDNA5) ----------------
#if USE_ASYNC_LDS
// GLOBAL_LOAD_ASYNC_TO_LDS_B128 (GV mode): per-lane 16B memory -> LDS.
// VDST VGPR carries the LDS byte address (low 32 bits of the flat LDS pointer).
__device__ __forceinline__ void async_copy16(void* lds_dst, const void* gsrc) {
    unsigned ldsa = (unsigned)(unsigned long long)lds_dst;
    asm volatile("global_load_async_to_lds_b128 %0, %1, off"
                 :: "v"(ldsa), "v"(gsrc) : "memory");
}
__device__ __forceinline__ void wait_async0() {
    asm volatile("s_wait_asynccnt 0" ::: "memory");
}
#else
__device__ __forceinline__ void async_copy16(void* lds_dst, const void* gsrc) {
    *(uint4*)lds_dst = *(const uint4*)gsrc;
}
__device__ __forceinline__ void wait_async0() {}
#endif

// ---------------- WMMA helpers ----------------

__device__ __forceinline__ v8f wmma_bf16(v16bf a, v16bf b, v8f c) {
    return __builtin_amdgcn_wmma_f32_16x16x32_bf16(
        /*neg_a=*/false, a, /*neg_b=*/false, b,
        /*c_mod=*/(short)0, c, /*reuse_a=*/false, /*reuse_b=*/false);
}

__device__ __forceinline__ v16bf frag_combine(v8bf lo, v8bf hi) {
    v16bf r;
#pragma unroll
    for (int i = 0; i < 8; ++i) { r[i] = lo[i]; r[i + 8] = hi[i]; }
    return r;
}

// A-operand fragment (16x32 bf16). LDS layout: row-major [rows][ld].
// Lane l: row = base_row + (l&15); K chunks {kc..kc+7, kc+16..kc+23}, kc = (l<16)?0:8.
__device__ __forceinline__ v16bf ld_frag_a(const __bf16* base, int row0, int ld,
                                           int k0, int lane) {
    int r  = row0 + (lane & 15);
    int kc = (lane < 16) ? 0 : 8;
    const __bf16* p = base + r * ld + k0 + kc;
    v8bf lo = *(const v8bf*)p;
    v8bf hi = *(const v8bf*)(p + 16);
    return frag_combine(lo, hi);
}

// B-operand fragment (32x16 bf16). LDS layout: "Bt" [cols][ld] (K contiguous per col).
// Lane l: col = base_col + (l&15); K chunks {kb..kb+7, kb+16..kb+23}, kb = (l<16)?0:8.
__device__ __forceinline__ v16bf ld_frag_b(const __bf16* base, int col0, int ld,
                                           int k0, int lane) {
    int c  = col0 + (lane & 15);
    int kb = (lane < 16) ? 0 : 8;
    const __bf16* p = base + c * ld + k0 + kb;
    v8bf lo = *(const v8bf*)p;
    v8bf hi = *(const v8bf*)(p + 16);
    return frag_combine(lo, hi);
}

__device__ __forceinline__ v8f zero8() {
    v8f z = {0.f, 0.f, 0.f, 0.f, 0.f, 0.f, 0.f, 0.f};
    return z;
}

// ---------------- Kernel 1: QKV GEMM (slots 0 and 2 only) ----------------
// C[16384, 1536] = x @ w_qkv[:, {0:768, 1536:2304}]
// n < 768  -> q_bf16 (scaled by SCALE);  n >= 768 -> k_bf16 and v_f32
__global__ __launch_bounds__(256) void gemm_qkv_kernel(
    const float* __restrict__ x, const float* __restrict__ wq,
    __bf16* __restrict__ q16, __bf16* __restrict__ k16, float* __restrict__ vf) {
    extern __shared__ char smem[];
    __bf16* As  = (__bf16*)smem;          // [128][32]
    __bf16* Bts = As + 128 * 32;          // [128 cols][32 k]

    const int tid  = threadIdx.x;
    const int lane = tid & 31;
    const int w    = tid >> 5;
    const int row0  = blockIdx.x * 128;
    const int ncol0 = blockIdx.y * 128;
    const int mbase = (w & 3) * 32;
    const int nbase = (w >> 2) * 64;

    v8f acc[2][4];
#pragma unroll
    for (int mi = 0; mi < 2; ++mi)
#pragma unroll
        for (int ni = 0; ni < 4; ++ni) acc[mi][ni] = zero8();

    for (int k0 = 0; k0 < DMODEL; k0 += 32) {
        // Stage A: x tile (f32 -> bf16), coalesced along k
        for (int i = tid; i < 128 * 32; i += 256) {
            int r = i >> 5, c = i & 31;
            As[i] = (__bf16)x[(size_t)(row0 + r) * DMODEL + k0 + c];
        }
        // Stage B transposed: Bts[n][c] = w_qkv[k0+c][col(n)], coalesced along n
        for (int i = tid; i < 128 * 32; i += 256) {
            int c = i >> 7, n = i & 127;
            int wc = ncol0 + n;
            if (wc >= 768) wc += 768;  // map to slot 2
            Bts[n * 32 + c] = (__bf16)wq[(size_t)(k0 + c) * 2304 + wc];
        }
        __syncthreads();

        v16bf af[2];
#pragma unroll
        for (int mi = 0; mi < 2; ++mi)
            af[mi] = ld_frag_a(As, mbase + mi * 16, 32, 0, lane);
#pragma unroll
        for (int ni = 0; ni < 4; ++ni) {
            v16bf bf_ = ld_frag_b(Bts, nbase + ni * 16, 32, 0, lane);
#pragma unroll
            for (int mi = 0; mi < 2; ++mi)
                acc[mi][ni] = wmma_bf16(af[mi], bf_, acc[mi][ni]);
        }
        __syncthreads();
    }

    const int hi8 = (lane < 16) ? 0 : 8;
    const int cl  = lane & 15;
#pragma unroll
    for (int mi = 0; mi < 2; ++mi)
#pragma unroll
        for (int ni = 0; ni < 4; ++ni)
#pragma unroll
            for (int i = 0; i < 8; ++i) {
                int r = row0 + mbase + mi * 16 + i + hi8;
                int n = ncol0 + nbase + ni * 16 + cl;
                float v = acc[mi][ni][i];
                if (n < 768) {
                    q16[(size_t)r * DMODEL + n] = (__bf16)(v * SCALE);
                } else {
                    k16[(size_t)r * DMODEL + (n - 768)] = (__bf16)v;
                    vf [(size_t)r * DMODEL + (n - 768)] = v;
                }
            }
}

// ---------------- Kernel 2: flash attention ----------------
// Per WG: 32 query rows, stream 64-key tiles, online softmax, O in registers.
__global__ __launch_bounds__(256, 1) void flash_attn_kernel(
    const __bf16* __restrict__ qg, const __bf16* __restrict__ kg,
    __bf16* __restrict__ og) {
    extern __shared__ char smem[];
    __bf16* qs  = (__bf16*)smem;                 // [QT][DMODEL]
    __bf16* ks  = qs + QT * DMODEL;              // [KT][DMODEL]  (row-major: QK^T B-op)
    __bf16* vts = ks + KT * DMODEL;              // [DMODEL][KT]  (transposed: PV B-op)
    float*  Ss  = (float*)(vts + DMODEL * KT);   // [QT][KT]
    __bf16* Ps  = (__bf16*)(Ss + QT * KT);       // [QT][KT]
    float*  ms  = (float*)(Ps + QT * KT);        // [QT] running max
    float*  ls  = ms + QT;                       // [QT] running sum
    float*  cs  = ls + QT;                       // [QT] correction factor

    const int tid  = threadIdx.x;
    const int lane = tid & 31;
    const int w    = tid >> 5;
    const int q0 = blockIdx.x * QT;
    const int b  = blockIdx.y;
    const __bf16* qgb = qg + ((size_t)b * SEQ + q0) * DMODEL;
    const __bf16* kgb = kg + (size_t)b * SEQ * DMODEL;

    // Load whole q tile once (bf16, 48 KB): async memory->LDS, 16B per op
    for (int i = tid; i < QT * DMODEL * 2 / 16; i += 256)
        async_copy16(qs + i * 8, qgb + i * 8);
    if (tid < QT) { ms[tid] = -3.0e38f; ls[tid] = 0.f; }

    const int mt = w & 1;    // M tile (rows 16*mt..)
    const int nq = w >> 1;   // 192-col quarter of O
    v8f acc[12];
#pragma unroll
    for (int j = 0; j < 12; ++j) acc[j] = zero8();

    wait_async0();
    __syncthreads();

    for (int kt = 0; kt < SEQ / KT; ++kt) {
        const __bf16* ksrc = kgb + (size_t)kt * KT * DMODEL;
        // Stage k row-major: async memory->LDS (96 KB / 24 ops per thread)
        for (int i = tid; i < KT * DMODEL * 2 / 16; i += 256)
            async_copy16(ks + i * 8, ksrc + i * 8);
        // Stage v transposed: vts[d][key]. Consecutive lanes take consecutive
        // keys -> LDS stores are address-consecutive (bank-conflict-free);
        // the scattered 16B global reads are served from L2 (k fits in 192MB).
        for (int i8 = tid; i8 < KT * DMODEL / 8; i8 += 256) {
            int key = i8 & (KT - 1);
            int d0  = (i8 >> 6) * 8;
            v8bf v = *(const v8bf*)(ksrc + (size_t)key * DMODEL + d0);
#pragma unroll
            for (int t = 0; t < 8; ++t) vts[(d0 + t) * KT + key] = v[t];
        }
        wait_async0();
        __syncthreads();

        // S = (q*SCALE) @ k^T : 8 waves x one 16x16 tile (2 M x 4 Kcol tiles)
        {
            v8f s = zero8();
#pragma unroll
            for (int k0 = 0; k0 < DMODEL; k0 += 32) {
                v16bf a  = ld_frag_a(qs, mt * 16, DMODEL, k0, lane);
                v16bf bb = ld_frag_b(ks, nq * 16, DMODEL, k0, lane);
                s = wmma_bf16(a, bb, s);
            }
            const int hi8 = (lane < 16) ? 0 : 8;
            const int cl  = lane & 15;
#pragma unroll
            for (int i = 0; i < 8; ++i)
                Ss[(mt * 16 + i + hi8) * KT + nq * 16 + cl] = s[i];
        }
        __syncthreads();

        // Online softmax: one thread per query row
        if (tid < QT) {
            float mold = ms[tid];
            float mnew = mold;
            float* srow = Ss + tid * KT;
#pragma unroll
            for (int j = 0; j < KT; ++j) mnew = fmaxf(mnew, srow[j]);
            float c = __expf(mold - mnew);
            float sum = 0.f;
            __bf16* prow = Ps + tid * KT;
#pragma unroll
            for (int j = 0; j < KT; ++j) {
                float p = __expf(srow[j] - mnew);
                prow[j] = (__bf16)p;
                sum += p;
            }
            ms[tid] = mnew;
            cs[tid] = c;
            ls[tid] = ls[tid] * c + sum;
        }
        __syncthreads();

        // Rescale O and accumulate P @ V
        {
            const int hi8 = (lane < 16) ? 0 : 8;
            float crow[8];
#pragma unroll
            for (int i = 0; i < 8; ++i) crow[i] = cs[mt * 16 + i + hi8];
            v16bf a0 = ld_frag_a(Ps, mt * 16, KT, 0,  lane);
            v16bf a1 = ld_frag_a(Ps, mt * 16, KT, 32, lane);
#pragma unroll
            for (int j = 0; j < 12; ++j) {
                int c0 = nq * 192 + j * 16;
#pragma unroll
                for (int i = 0; i < 8; ++i) acc[j][i] *= crow[i];
                v16bf b0 = ld_frag_b(vts, c0, KT, 0,  lane);
                v16bf b1 = ld_frag_b(vts, c0, KT, 32, lane);
                acc[j] = wmma_bf16(a0, b0, acc[j]);
                acc[j] = wmma_bf16(a1, b1, acc[j]);
            }
        }
        __syncthreads();
    }

    // Epilogue: O /= l, write bf16
    const int hi8 = (lane < 16) ? 0 : 8;
    const int cl  = lane & 15;
    float rinv[8];
#pragma unroll
    for (int i = 0; i < 8; ++i) rinv[i] = 1.0f / ls[mt * 16 + i + hi8];
    __bf16* ogb = og + ((size_t)b * SEQ + q0) * DMODEL;
#pragma unroll
    for (int j = 0; j < 12; ++j) {
        int c0 = nq * 192 + j * 16 + cl;
#pragma unroll
        for (int i = 0; i < 8; ++i)
            ogb[(size_t)(mt * 16 + i + hi8) * DMODEL + c0] = (__bf16)(acc[j][i] * rinv[i]);
    }
}

// ---------------- Kernel 3: projection + bias + residual ----------------
// out = o @ w_proj + b_proj + v
__global__ __launch_bounds__(256) void gemm_proj_kernel(
    const __bf16* __restrict__ o16, const float* __restrict__ wp,
    const float* __restrict__ bp, const float* __restrict__ vf,
    float* __restrict__ out) {
    extern __shared__ char smem[];
    __bf16* As  = (__bf16*)smem;   // [128][32]
    __bf16* Bts = As + 128 * 32;   // [128 cols][32 k]

    const int tid  = threadIdx.x;
    const int lane = tid & 31;
    const int w    = tid >> 5;
    const int row0  = blockIdx.x * 128;
    const int ncol0 = blockIdx.y * 128;
    const int mbase = (w & 3) * 32;
    const int nbase = (w >> 2) * 64;

    v8f acc[2][4];
#pragma unroll
    for (int mi = 0; mi < 2; ++mi)
#pragma unroll
        for (int ni = 0; ni < 4; ++ni) acc[mi][ni] = zero8();

    for (int k0 = 0; k0 < DMODEL; k0 += 32) {
        // Stage A: bf16 tile copy via async memory->LDS (4 x 16B per row)
        for (int i = tid; i < 128 * 4; i += 256) {
            int r = i >> 2, c16 = i & 3;
            async_copy16(As + r * 32 + c16 * 8,
                         o16 + (size_t)(row0 + r) * DMODEL + k0 + c16 * 8);
        }
        for (int i = tid; i < 128 * 32; i += 256) {
            int c = i >> 7, n = i & 127;
            Bts[n * 32 + c] = (__bf16)wp[(size_t)(k0 + c) * DMODEL + ncol0 + n];
        }
        wait_async0();
        __syncthreads();

        v16bf af[2];
#pragma unroll
        for (int mi = 0; mi < 2; ++mi)
            af[mi] = ld_frag_a(As, mbase + mi * 16, 32, 0, lane);
#pragma unroll
        for (int ni = 0; ni < 4; ++ni) {
            v16bf bf_ = ld_frag_b(Bts, nbase + ni * 16, 32, 0, lane);
#pragma unroll
            for (int mi = 0; mi < 2; ++mi)
                acc[mi][ni] = wmma_bf16(af[mi], bf_, acc[mi][ni]);
        }
        __syncthreads();
    }

    const int hi8 = (lane < 16) ? 0 : 8;
    const int cl  = lane & 15;
#pragma unroll
    for (int mi = 0; mi < 2; ++mi)
#pragma unroll
        for (int ni = 0; ni < 4; ++ni)
#pragma unroll
            for (int i = 0; i < 8; ++i) {
                int r = row0 + mbase + mi * 16 + i + hi8;
                int n = ncol0 + nbase + ni * 16 + cl;
                out[(size_t)r * DMODEL + n] =
                    acc[mi][ni][i] + bp[n] + vf[(size_t)r * DMODEL + n];
            }
}

// ---------------- Host launcher ----------------
extern "C" void kernel_launch(void* const* d_in, const int* in_sizes, int n_in,
                              void* d_out, int out_size, void* d_ws, size_t ws_size,
                              hipStream_t stream) {
    (void)in_sizes; (void)n_in; (void)out_size; (void)ws_size;
    const float* x  = (const float*)d_in[0];
    const float* wq = (const float*)d_in[1];
    const float* wp = (const float*)d_in[2];
    const float* bp = (const float*)d_in[3];
    float* out = (float*)d_out;

    char* ws = (char*)d_ws;
    const size_t rowsD = (size_t)4 * SEQ * DMODEL;   // 16384 * 768
    __bf16* q16 = (__bf16*)(ws);                     // 24 MB
    __bf16* k16 = (__bf16*)(ws + rowsD * 2);         // 24 MB (also v in bf16)
    float*  vf  = (float*) (ws + rowsD * 4);         // 48 MB (f32 residual)
    __bf16* o16 = (__bf16*)(ws + rowsD * 8);         // 24 MB

    const size_t gemm_smem = (size_t)2 * 128 * 32 * sizeof(__bf16); // 16 KB
    const size_t flash_smem =
        (size_t)(QT * DMODEL + KT * DMODEL + DMODEL * KT) * 2 + // q, k, v^T
        (size_t)QT * KT * 4 +                                   // S f32
        (size_t)QT * KT * 2 +                                   // P bf16
        (size_t)3 * QT * 4;                                     // m, l, c

    gemm_qkv_kernel<<<dim3(128, 12), 256, gemm_smem, stream>>>(x, wq, q16, k16, vf);
    flash_attn_kernel<<<dim3(SEQ / QT, 4), 256, flash_smem, stream>>>(q16, k16, o16);
    gemm_proj_kernel<<<dim3(128, 6), 256, gemm_smem, stream>>>(o16, wp, bp, vf, out);
}
